// RTGU_33603824124517
// MI455X (gfx1250) — matrix operand order repaired and checked by reading, not compiled, not run on hardware
//
#include <hip/hip_runtime.h>
#include <stdint.h>

// ---------------------------------------------------------------------------
// 2-layer GRU on MI455X (gfx1250):
//   - bf16 WMMA (v_wmma_f32_16x16x32_bf16) for all GEMMs, f32 accumulate
//   - async global->LDS staging (global_load_async_to_lds_b128 / ASYNCcnt)
//   - persistent scan kernel with grid barrier (2 barriers / timestep)
// ---------------------------------------------------------------------------

#define DEV __device__ __forceinline__

typedef __attribute__((ext_vector_type(16))) __bf16 v16bf;
typedef __attribute__((ext_vector_type(8)))  float  v8f;

static constexpr int Bsz = 64, Tn = 1024, En = 256, Hn = 512, Cn = 2;
static constexpr int Mrows = Bsz * Tn;       // 65536 rows for time-parallel GEMMs
static constexpr int SCAN_BLOCKS = 64;       // 64 blocks * 4 waves = 256 waves
static constexpr int LDSK = Hn + 8;          // padded LDS row pitch (bank spread)

union FragU { uint4 u[2]; v16bf v; };

DEV unsigned short f2bf(float x) {
  unsigned u = __builtin_bit_cast(unsigned, x);
  unsigned r = u + 0x7FFFu + ((u >> 16) & 1u);   // round-to-nearest-even
  return (unsigned short)(r >> 16);
}

// ISA 10.2: for an LDS flat address, ADDR[31:0] is the allocation-relative LDS
// byte address -- exactly what async-to-LDS instructions take as dsaddr.
DEV unsigned lds_off(const void* p) { return (unsigned)(uintptr_t)p; }

// Async DMA of one 16-byte chunk per lane, global -> LDS (tracked by ASYNCcnt).
DEV void async_g2l_b128(unsigned ldsByteOff, const unsigned short* gptr) {
  asm volatile("global_load_async_to_lds_b128 %0, %1, off"
               :: "v"(ldsByteOff), "v"(gptr) : "memory");
}
DEV void wait_async0() { asm volatile("s_wait_asynccnt 0x0" ::: "memory"); }

// A fragment (16x32 bf16, MxK): lane L -> row m=L&15, hi=L>>4,
// element i -> k = 8*hi + (i&7) + (i>=8 ? 16 : 0). Two 16B loads per lane.
DEV v16bf load_a_frag(const unsigned short* A, int ld, int row0, int k0, int lane) {
  int m  = row0 + (lane & 15);
  int hi = lane >> 4;
  const unsigned short* p = A + (size_t)m * ld + (size_t)(k0 + 8 * hi);
  FragU f;
  f.u[0] = *(const uint4*)(p);
  f.u[1] = *(const uint4*)(p + 16);
  return f.v;
}

// B fragment (32x16 bf16, KxN) with B[k][n] = W[n][k], W row-major [N][K]:
// lane L -> column n=L&15, hi=L>>4, element i -> k = 8*hi + (i&7) + (i>=8 ? 16 : 0).
DEV v16bf load_b_frag(const unsigned short* W, int ld, int n0, int k0, int lane) {
  int n  = n0 + (lane & 15);
  int hi = lane >> 4;
  const unsigned short* p = W + (size_t)n * ld + (size_t)(k0 + 8 * hi);
  FragU f;
  f.u[0] = *(const uint4*)(p);
  f.u[1] = *(const uint4*)(p + 16);
  return f.v;
}

DEV v8f wmma_bf16(v16bf a, v16bf b, v8f c) {
  return __builtin_amdgcn_wmma_f32_16x16x32_bf16(false, a, false, b, (short)0, c,
                                                 false, false);
}

DEV float fsigmoid(float x) { return 1.0f / (1.0f + __expf(-x)); }

// ---------------------------------------------------------------------------
// Grid-wide barrier (workspace counters, monotonically increasing generation)
// ---------------------------------------------------------------------------
DEV void grid_barrier(unsigned* cnt, unsigned* gen, int nblocks) {
  __syncthreads();
  if (threadIdx.x == 0) {
    __threadfence();
    unsigned old_gen = __hip_atomic_load(gen, __ATOMIC_ACQUIRE, __HIP_MEMORY_SCOPE_AGENT);
    unsigned arrived = __hip_atomic_fetch_add(cnt, 1u, __ATOMIC_ACQ_REL, __HIP_MEMORY_SCOPE_AGENT);
    if (arrived == (unsigned)(nblocks - 1)) {
      __hip_atomic_store(cnt, 0u, __ATOMIC_RELAXED, __HIP_MEMORY_SCOPE_AGENT);
      __hip_atomic_fetch_add(gen, 1u, __ATOMIC_ACQ_REL, __HIP_MEMORY_SCOPE_AGENT);
    } else {
      while (__hip_atomic_load(gen, __ATOMIC_ACQUIRE, __HIP_MEMORY_SCOPE_AGENT) == old_gen) {
        __builtin_amdgcn_s_sleep(2);
      }
    }
    __threadfence();
  }
  __syncthreads();
}

// ---------------------------------------------------------------------------
// Small helper kernels
// ---------------------------------------------------------------------------
__global__ void bar_init_kernel(unsigned* cnt, unsigned* gen) {
  if (threadIdx.x == 0) { *cnt = 0u; *gen = 0u; }
}

__global__ void cvt_slice_kernel(const float* __restrict__ src,
                                 unsigned short* __restrict__ dst,
                                 int rows, int src_ld, int col0, int cols) {
  int i = blockIdx.x * blockDim.x + threadIdx.x;
  if (i >= rows * cols) return;
  int r = i / cols, c = i - r * cols;
  dst[(size_t)r * cols + c] = f2bf(src[(size_t)r * src_ld + col0 + c]);
}

__global__ void embed_kernel(const int* __restrict__ tokens,
                             const float* __restrict__ emb,
                             unsigned short* __restrict__ X) {
  int r = blockIdx.x;
  int k = threadIdx.x;
  int tok = tokens[r];
  X[(size_t)r * En + k] = f2bf(emb[(size_t)tok * En + k]);
}

// ---------------------------------------------------------------------------
// Input-projection GEMM: Out[M][Hn] = A[M][K](bf16) @ W[Hn][K]^T + bias.
// grid = (Hn/64, M/128), block = 256 (8 waves). All waves share the 64-column
// W strip -> staged once into LDS via async DMA; A streams from global.
// ---------------------------------------------------------------------------
__global__ __launch_bounds__(256) void gemm_bias_kernel(
    const unsigned short* __restrict__ A, int K,
    const unsigned short* __restrict__ W,
    const float* __restrict__ bias,
    float* __restrict__ Out) {
  __shared__ unsigned short Bs[64 * LDSK];     // 65 KB weight strip

  int lane = threadIdx.x & 31;
  int wave = threadIdx.x >> 5;
  int row0 = (blockIdx.y * 8 + wave) * 16;
  int n0   = blockIdx.x * 64;

  // stage W rows [n0, n0+64) x K into LDS (16B chunks, one per lane issue)
  int chunksPerRow = K >> 3;
  int total = 64 * chunksPerRow;
  for (int i = threadIdx.x; i < total; i += 256) {
    int r = i / chunksPerRow, c = i - r * chunksPerRow;
    async_g2l_b128(lds_off(&Bs[r * LDSK + c * 8]),
                   W + (size_t)(n0 + r) * K + c * 8);
  }
  wait_async0();
  __syncthreads();

  v8f acc0 = {}, acc1 = {}, acc2 = {}, acc3 = {};
  for (int k0 = 0; k0 < K; k0 += 32) {
    v16bf a = load_a_frag(A, K, row0, k0, lane);
    if (k0 + 64 < K)
      __builtin_prefetch(A + (size_t)row0 * K + k0 + 64, 0, 1);
    v16bf b0 = load_b_frag(Bs, LDSK,  0, k0, lane);
    v16bf b1 = load_b_frag(Bs, LDSK, 16, k0, lane);
    v16bf b2 = load_b_frag(Bs, LDSK, 32, k0, lane);
    v16bf b3 = load_b_frag(Bs, LDSK, 48, k0, lane);
    acc0 = wmma_bf16(a, b0, acc0);
    acc1 = wmma_bf16(a, b1, acc1);
    acc2 = wmma_bf16(a, b2, acc2);
    acc3 = wmma_bf16(a, b3, acc3);
  }

  int hi = lane >> 4, nl = lane & 15;
  v8f accs[4] = {acc0, acc1, acc2, acc3};
#pragma unroll
  for (int j = 0; j < 4; ++j) {
    int n = n0 + 16 * j + nl;
    float bn = bias[n];
#pragma unroll
    for (int r = 0; r < 8; ++r) {
      int m = row0 + r + 8 * hi;
      Out[(size_t)m * Hn + n] = accs[j][r] + bn;
    }
  }
}

// ---------------------------------------------------------------------------
// Persistent GRU scan. 64 blocks x 128 threads (4 waves). Block b:
//   gate g = b>>5 (0:r, 1:z), batch tile mt1 = (b>>3)&3, strip s = b&7
//   phase1: wave j computes 16x16 tile at (mt1*16, s*64 + j*16) of gate g
//   phase2 (b<32): mt2 = b>>3, same strip, Whh
// Weight strips (loop-invariant!) staged to LDS once via async DMA; the
// 16-row h / (r.*h) slab is staged per phase. All WMMA operands read from LDS.
// ---------------------------------------------------------------------------
__global__ __launch_bounds__(128) void gru_scan_kernel(
    const float* __restrict__ gr, const float* __restrict__ gz,
    const float* __restrict__ gh,
    const unsigned short* __restrict__ Wrh,
    const unsigned short* __restrict__ Wzh,
    const unsigned short* __restrict__ Whh,
    float* __restrict__ h_f32, unsigned short* __restrict__ h_bf,
    unsigned short* __restrict__ rh_bf, float* __restrict__ z_f32,
    unsigned short* __restrict__ hs_out,       // null for layer 1
    unsigned* bar_cnt, unsigned* bar_gen, int T, int nblocks) {
  __shared__ unsigned short wA[64 * LDSK];     // phase1 gate-weight strip (65 KB)
  __shared__ unsigned short wB[64 * LDSK];     // phase2 Whh strip (65 KB)
  __shared__ unsigned short xs[16 * LDSK];     // h / (r.*h) slab (16.6 KB)

  int tid   = threadIdx.x;
  int lane  = tid & 31;
  int j     = tid >> 5;                        // wave in block, 0..3
  int b     = blockIdx.x;
  int g     = b >> 5;                          // 0 = r-gate, 1 = z-gate
  int mt1   = (b >> 3) & 3;                    // phase1 batch tile
  int mt2   = b >> 3;                          // phase2 batch tile (b<32)
  int nbase = (b & 7) * 64;                    // hidden strip origin

  // ---- one-time LDS staging of the weight strips ----
  const unsigned short* WgA = g ? Wzh : Wrh;
  for (int i = tid; i < 64 * 64; i += 128) {   // 64 rows x 64 16B-chunks
    int r = i >> 6, c = i & 63;
    async_g2l_b128(lds_off(&wA[r * LDSK + c * 8]),
                   WgA + (size_t)(nbase + r) * Hn + c * 8);
  }
  if (b < 32) {
    for (int i = tid; i < 64 * 64; i += 128) {
      int r = i >> 6, c = i & 63;
      async_g2l_b128(lds_off(&wB[r * LDSK + c * 8]),
                     Whh + (size_t)(nbase + r) * Hn + c * 8);
    }
  }
  wait_async0();

  // zero-init hidden state (workspace is poisoned by the harness)
  for (int i = b * 128 + tid; i < Bsz * Hn; i += nblocks * 128) {
    h_f32[i] = 0.0f;
    h_bf[i]  = 0;
  }
  grid_barrier(bar_cnt, bar_gen, nblocks);

  for (int t = 0; t < T; ++t) {
    // ---- stage h slab rows [mt1*16, +16) into LDS ----
    for (int i = tid; i < 16 * 64; i += 128) {
      int r = i >> 6, c = i & 63;
      async_g2l_b128(lds_off(&xs[r * LDSK + c * 8]),
                     h_bf + (size_t)(mt1 * 16 + r) * Hn + c * 8);
    }
    wait_async0();
    __syncthreads();

    { // ---- phase 1: r / z gate tile ----
      v8f acc = {};
      for (int k0 = 0; k0 < Hn; k0 += 32) {
        v16bf a  = load_a_frag(xs, LDSK, 0, k0, lane);
        v16bf bf = load_b_frag(wA, LDSK, j * 16, k0, lane);
        acc = wmma_bf16(a, bf, acc);
      }
      int hi = lane >> 4, nl = lane & 15;
      int n = nbase + j * 16 + nl;
      const float* plane = g ? gz : gr;
#pragma unroll
      for (int r = 0; r < 8; ++r) {
        int m = mt1 * 16 + r + 8 * hi;                 // batch index
        float pre = plane[((size_t)m * T + t) * Hn + n];
        float v = fsigmoid(pre + acc[r]);
        size_t idx = (size_t)m * Hn + n;
        if (g == 0) rh_bf[idx] = f2bf(v * h_f32[idx]); // (r .* h) for phase 2
        else        z_f32[idx] = v;
      }
    }
    grid_barrier(bar_cnt, bar_gen, nblocks);

    if (b < 32) { // ---- phase 2: candidate + gated update ----
      for (int i = tid; i < 16 * 64; i += 128) {
        int r = i >> 6, c = i & 63;
        async_g2l_b128(lds_off(&xs[r * LDSK + c * 8]),
                       rh_bf + (size_t)(mt2 * 16 + r) * Hn + c * 8);
      }
      wait_async0();
      __syncthreads();

      v8f acc = {};
      for (int k0 = 0; k0 < Hn; k0 += 32) {
        v16bf a  = load_a_frag(xs, LDSK, 0, k0, lane);
        v16bf bf = load_b_frag(wB, LDSK, j * 16, k0, lane);
        acc = wmma_bf16(a, bf, acc);
      }
      int hi = lane >> 4, nl = lane & 15;
      int n = nbase + j * 16 + nl;
#pragma unroll
      for (int r = 0; r < 8; ++r) {
        int m = mt2 * 16 + r + 8 * hi;
        size_t idx = (size_t)m * Hn + n;
        float pre = gh[((size_t)m * T + t) * Hn + n];
        float ht  = tanhf(pre + acc[r]);
        float z   = z_f32[idx];
        float hn  = (1.0f - z) * h_f32[idx] + z * ht;
        h_f32[idx] = hn;
        unsigned short hb = f2bf(hn);
        h_bf[idx] = hb;
        if (hs_out) hs_out[((size_t)m * T + t) * Hn + n] = hb;  // layer-1 input
      }
    }
    grid_barrier(bar_cnt, bar_gen, nblocks);
  }
}

// ---------------------------------------------------------------------------
// Final FC: out[64][2] = h_last @ fcW^T + fcb  (tiny, VALU)
// ---------------------------------------------------------------------------
__global__ void fc_kernel(const float* __restrict__ h, const float* __restrict__ W,
                          const float* __restrict__ b, float* __restrict__ out) {
  int i = threadIdx.x;
  if (i >= Bsz * Cn) return;
  int bb = i >> 1, c = i & 1;
  float s = 0.0f;
  for (int jj = 0; jj < Hn; ++jj) s += h[bb * Hn + jj] * W[c * Hn + jj];
  out[i] = s + b[c];
}

// ---------------------------------------------------------------------------
// Host driver
// ---------------------------------------------------------------------------
extern "C" void kernel_launch(void* const* d_in, const int* in_sizes, int n_in,
                              void* d_out, int out_size, void* d_ws, size_t ws_size,
                              hipStream_t stream) {
  (void)in_sizes; (void)n_in; (void)out_size; (void)ws_size;

  const int*   tokens = (const int*)  d_in[0];
  const float* emb    = (const float*)d_in[1];
  const float* Wr0 = (const float*)d_in[2];  const float* br0 = (const float*)d_in[3];
  const float* Wz0 = (const float*)d_in[4];  const float* bz0 = (const float*)d_in[5];
  const float* Wh0 = (const float*)d_in[6];  const float* bh0 = (const float*)d_in[7];
  const float* Wr1 = (const float*)d_in[8];  const float* br1 = (const float*)d_in[9];
  const float* Wz1 = (const float*)d_in[10]; const float* bz1 = (const float*)d_in[11];
  const float* Wh1 = (const float*)d_in[12]; const float* bh1 = (const float*)d_in[13];
  const float* fcW = (const float*)d_in[14]; const float* fcb = (const float*)d_in[15];

  char* p = (char*)d_ws;
  auto carve = [&](size_t bytes) -> void* {
    void* r = (void*)p;
    p += (bytes + 255) & ~(size_t)255;
    return r;
  };

  unsigned short* X0  = (unsigned short*)carve((size_t)Mrows * En * 2);   // 32 MB
  float* g_r = (float*)carve((size_t)Mrows * Hn * 4);                     // 3x128 MB,
  float* g_z = (float*)carve((size_t)Mrows * Hn * 4);                     // reused by
  float* g_h = (float*)carve((size_t)Mrows * Hn * 4);                     // both layers
  unsigned short* hs0 = (unsigned short*)carve((size_t)Mrows * Hn * 2);   // 64 MB
  unsigned short* Wrx0 = (unsigned short*)carve((size_t)Hn * En * 2);
  unsigned short* Wzx0 = (unsigned short*)carve((size_t)Hn * En * 2);
  unsigned short* Whx0 = (unsigned short*)carve((size_t)Hn * En * 2);
  unsigned short* Wrh0 = (unsigned short*)carve((size_t)Hn * Hn * 2);
  unsigned short* Wzh0 = (unsigned short*)carve((size_t)Hn * Hn * 2);
  unsigned short* Whh0 = (unsigned short*)carve((size_t)Hn * Hn * 2);
  unsigned short* Wrx1 = (unsigned short*)carve((size_t)Hn * Hn * 2);
  unsigned short* Wzx1 = (unsigned short*)carve((size_t)Hn * Hn * 2);
  unsigned short* Whx1 = (unsigned short*)carve((size_t)Hn * Hn * 2);
  unsigned short* Wrh1 = (unsigned short*)carve((size_t)Hn * Hn * 2);
  unsigned short* Wzh1 = (unsigned short*)carve((size_t)Hn * Hn * 2);
  unsigned short* Whh1 = (unsigned short*)carve((size_t)Hn * Hn * 2);
  float*          h_f32 = (float*)carve((size_t)Bsz * Hn * 4);
  unsigned short* h_bf  = (unsigned short*)carve((size_t)Bsz * Hn * 2);
  unsigned short* rh_bf = (unsigned short*)carve((size_t)Bsz * Hn * 2);
  float*          z_f32 = (float*)carve((size_t)Bsz * Hn * 4);
  unsigned*       bar   = (unsigned*)carve(64);

  auto cvt = [&](const float* src, unsigned short* dst, int rows, int src_ld,
                 int col0, int cols) {
    int total = rows * cols;
    cvt_slice_kernel<<<(total + 255) / 256, 256, 0, stream>>>(src, dst, rows,
                                                              src_ld, col0, cols);
  };
  cvt(Wr0, Wrx0, Hn, En + Hn, 0, En);  cvt(Wr0, Wrh0, Hn, En + Hn, En, Hn);
  cvt(Wz0, Wzx0, Hn, En + Hn, 0, En);  cvt(Wz0, Wzh0, Hn, En + Hn, En, Hn);
  cvt(Wh0, Whx0, Hn, En + Hn, 0, En);  cvt(Wh0, Whh0, Hn, En + Hn, En, Hn);
  cvt(Wr1, Wrx1, Hn, Hn + Hn, 0, Hn);  cvt(Wr1, Wrh1, Hn, Hn + Hn, Hn, Hn);
  cvt(Wz1, Wzx1, Hn, Hn + Hn, 0, Hn);  cvt(Wz1, Wzh1, Hn, Hn + Hn, Hn, Hn);
  cvt(Wh1, Whx1, Hn, Hn + Hn, 0, Hn);  cvt(Wh1, Whh1, Hn, Hn + Hn, Hn, Hn);

  embed_kernel<<<Mrows, En, 0, stream>>>(tokens, emb, X0);

  dim3 ggrid(Hn / 64, Mrows / 128);

  // ---- layer 0: input projections + scan ----
  gemm_bias_kernel<<<ggrid, 256, 0, stream>>>(X0, En, Wrx0, br0, g_r);
  gemm_bias_kernel<<<ggrid, 256, 0, stream>>>(X0, En, Wzx0, bz0, g_z);
  gemm_bias_kernel<<<ggrid, 256, 0, stream>>>(X0, En, Whx0, bh0, g_h);
  bar_init_kernel<<<1, 32, 0, stream>>>(bar, bar + 1);
  gru_scan_kernel<<<SCAN_BLOCKS, 128, 0, stream>>>(
      g_r, g_z, g_h, Wrh0, Wzh0, Whh0, h_f32, h_bf, rh_bf, z_f32, hs0,
      bar, bar + 1, Tn, SCAN_BLOCKS);

  // ---- layer 1: input projections (gate planes reused) + scan ----
  gemm_bias_kernel<<<ggrid, 256, 0, stream>>>(hs0, Hn, Wrx1, br1, g_r);
  gemm_bias_kernel<<<ggrid, 256, 0, stream>>>(hs0, Hn, Wzx1, bz1, g_z);
  gemm_bias_kernel<<<ggrid, 256, 0, stream>>>(hs0, Hn, Whx1, bh1, g_h);
  bar_init_kernel<<<1, 32, 0, stream>>>(bar, bar + 1);
  gru_scan_kernel<<<SCAN_BLOCKS, 128, 0, stream>>>(
      g_r, g_z, g_h, Wrh1, Wzh1, Whh1, h_f32, h_bf, rh_bf, z_f32,
      (unsigned short*)nullptr, bar, bar + 1, Tn, SCAN_BLOCKS);

  // ---- classifier head ----
  fc_kernel<<<1, 128, 0, stream>>>(h_f32, fcW, fcb, (float*)d_out);
}
